// GaussianMLP_18236431139420
// MI455X (gfx1250) — matrix operand order repaired and checked by reading, not compiled, not run on hardware
//
#include <hip/hip_runtime.h>
#include <hip/hip_bf16.h>

typedef __attribute__((ext_vector_type(16))) _Float16 v16h;
typedef __attribute__((ext_vector_type(8)))  _Float16 v8h;
typedef __attribute__((ext_vector_type(8)))  float    v8f;
typedef __attribute__((ext_vector_type(4)))  float    v4f;

#define D_IN   128
#define D_H    256
#define D_OUT  128
#define WAVES  8
#define ROWS_PER_WAVE  32                         // two 16-row M-tiles per wave
#define ROWS_PER_BLOCK (WAVES * ROWS_PER_WAVE)    // 256 batch rows per block

__global__ __launch_bounds__(256, 1)
void GaussianMLP_vae_enc_kernel(const float* __restrict__ x,
                                const float* __restrict__ eps,
                                const float* __restrict__ W_emb,
                                const float* __restrict__ b_emb,
                                const float* __restrict__ W_mean,
                                const float* __restrict__ b_mean,
                                const float* __restrict__ W_logvar,
                                const float* __restrict__ b_logvar,
                                float* __restrict__ z_out,
                                float* __restrict__ mean_out,
                                float* __restrict__ logvar_out,
                                int batch)
{
    // f16 weights, transposed to [N][K] so WMMA B-fragments are contiguous K runs.
    __shared__ __align__(16) _Float16 sWemb [D_H  * D_IN];          // 64 KB
    __shared__ __align__(16) _Float16 sWmean[D_OUT * D_H ];         // 64 KB
    __shared__ __align__(16) _Float16 sWlog [D_OUT * D_H ];         // 64 KB
    __shared__ __align__(16) _Float16 sH    [ROWS_PER_BLOCK * D_H]; // 128 KB
    // total static LDS = 320 KB (gfx1250 per-workgroup max)

    const int tid = threadIdx.x;

    // ---- Cooperative weight convert fp32 -> f16 (transposed) ----
    for (int i = tid; i < D_IN * D_H; i += 256) {
        int k = i / D_H, n = i % D_H;
        sWemb[n * D_IN + k] = (_Float16)W_emb[i];
    }
    for (int i = tid; i < D_H * D_OUT; i += 256) {
        int k = i / D_OUT, n = i % D_OUT;
        sWmean[n * D_H + k] = (_Float16)W_mean[i];
        sWlog [n * D_H + k] = (_Float16)W_logvar[i];
    }
    __syncthreads();

    const int wave  = tid >> 5;          // wave32
    const int lane  = tid & 31;
    const int lhalf = lane >> 4;         // 0: lanes 0-15, 1: lanes 16-31
    const int l16   = lane & 15;
    const long rowBase = (long)blockIdx.x * ROWS_PER_BLOCK + wave * ROWS_PER_WAVE;

    // =======================================================================
    // GEMM1: h = relu(x @ W_emb + b_emb), two 16-row M-tiles per wave.
    // A layout (16-bit, 16x32): halves[0..7]=K kb..kb+7, halves[8..15]=K kb+16..kb+23,
    //                           kb = kstep*32 + lhalf*8, row M = lane%16.
    // =======================================================================
    v16h afrag[2][4];
#pragma unroll
    for (int t = 0; t < 2; ++t) {
        const long xbase = (rowBase + t * 16 + l16) * (long)D_IN;
#pragma unroll
        for (int ks = 0; ks < 4; ++ks) {
            const int kb = ks * 32 + lhalf * 8;
            v4f x0 = *(const v4f*)(x + xbase + kb);
            v4f x1 = *(const v4f*)(x + xbase + kb + 4);
            v4f x2 = *(const v4f*)(x + xbase + kb + 16);
            v4f x3 = *(const v4f*)(x + xbase + kb + 20);
            v16h a;
#pragma unroll
            for (int j = 0; j < 4; ++j) {
                a[j]      = (_Float16)x0[j];
                a[4 + j]  = (_Float16)x1[j];
                a[8 + j]  = (_Float16)x2[j];
                a[12 + j] = (_Float16)x3[j];
            }
            afrag[t][ks] = a;
        }
    }

    _Float16* hrow = &sH[wave * ROWS_PER_WAVE * D_H];

#pragma unroll 4
    for (int nt = 0; nt < D_H / 16; ++nt) {
        const int ncol = nt * 16 + l16;
        v8f acc0 = {};
        v8f acc1 = {};
#pragma unroll
        for (int ks = 0; ks < 4; ++ks) {
            // B layout (16-bit, 32x16): column N = lane%16,
            // halves[0..7] = K lhalf*8..lhalf*8+7, halves[8..15] = K 16+lhalf*8..16+lhalf*8+7.
            const _Float16* bp = &sWemb[ncol * D_IN + ks * 32 + lhalf * 8];
            v8h b0 = *(const v8h*)bp;
            v8h b1 = *(const v8h*)(bp + 16);
            v16h b;
#pragma unroll
            for (int j = 0; j < 8; ++j) { b[j] = b0[j]; b[8 + j] = b1[j]; }
            // one B fragment feeds both M-tiles
            acc0 = __builtin_amdgcn_wmma_f32_16x16x32_f16(
                false, afrag[0][ks], false, b, (short)0, acc0, false, false);
            acc1 = __builtin_amdgcn_wmma_f32_16x16x32_f16(
                false, afrag[1][ks], false, b, (short)0, acc1, false, false);
        }
        const float bias = b_emb[ncol];
        // C/D layout: VGPR v -> row v + lhalf*8, col = ncol.
#pragma unroll
        for (int v = 0; v < 8; ++v) {
            float h0 = acc0[v] + bias;
            float h1 = acc1[v] + bias;
            h0 = h0 > 0.0f ? h0 : 0.0f;
            h1 = h1 > 0.0f ? h1 : 0.0f;
            hrow[(v + lhalf * 8) * D_H + ncol]        = (_Float16)h0;
            hrow[(16 + v + lhalf * 8) * D_H + ncol]   = (_Float16)h1;
        }
    }
    __syncthreads();   // order h stores vs. h A-fragment reloads

    // =======================================================================
    // GEMM2/3: mean/logvar = h @ W_{mean,logvar} + b, two M-tiles per wave.
    // =======================================================================
    v16h hfrag[2][8];
#pragma unroll
    for (int t = 0; t < 2; ++t) {
#pragma unroll
        for (int ks = 0; ks < 8; ++ks) {
            const _Float16* hp = &hrow[(t * 16 + l16) * D_H + ks * 32 + lhalf * 8];
            v8h h0 = *(const v8h*)hp;
            v8h h1 = *(const v8h*)(hp + 16);
            v16h a;
#pragma unroll
            for (int j = 0; j < 8; ++j) { a[j] = h0[j]; a[8 + j] = h1[j]; }
            hfrag[t][ks] = a;
        }
    }

#pragma unroll 2
    for (int nt = 0; nt < D_OUT / 16; ++nt) {
        const int ncol = nt * 16 + l16;
        v8f accm0 = {}, accm1 = {};
        v8f accl0 = {}, accl1 = {};
#pragma unroll
        for (int ks = 0; ks < 8; ++ks) {
            const _Float16* mp = &sWmean[ncol * D_H + ks * 32 + lhalf * 8];
            v8h m0 = *(const v8h*)mp;
            v8h m1 = *(const v8h*)(mp + 16);
            v16h bm;
#pragma unroll
            for (int j = 0; j < 8; ++j) { bm[j] = m0[j]; bm[8 + j] = m1[j]; }
            accm0 = __builtin_amdgcn_wmma_f32_16x16x32_f16(
                false, hfrag[0][ks], false, bm, (short)0, accm0, false, false);
            accm1 = __builtin_amdgcn_wmma_f32_16x16x32_f16(
                false, hfrag[1][ks], false, bm, (short)0, accm1, false, false);

            const _Float16* lp = &sWlog[ncol * D_H + ks * 32 + lhalf * 8];
            v8h l0 = *(const v8h*)lp;
            v8h l1 = *(const v8h*)(lp + 16);
            v16h bl;
#pragma unroll
            for (int j = 0; j < 8; ++j) { bl[j] = l0[j]; bl[8 + j] = l1[j]; }
            accl0 = __builtin_amdgcn_wmma_f32_16x16x32_f16(
                false, hfrag[0][ks], false, bl, (short)0, accl0, false, false);
            accl1 = __builtin_amdgcn_wmma_f32_16x16x32_f16(
                false, hfrag[1][ks], false, bl, (short)0, accl1, false, false);
        }
        const float bm_s = b_mean[ncol];
        const float bl_s = b_logvar[ncol];
        // Fused rsample epilogue: z = mean + exp(0.5*logvar) * eps
#pragma unroll
        for (int t = 0; t < 2; ++t) {
            const v8f& am = t ? accm1 : accm0;
            const v8f& al = t ? accl1 : accl0;
#pragma unroll
            for (int v = 0; v < 8; ++v) {
                const long row = rowBase + t * 16 + v + lhalf * 8;
                const long idx = row * D_OUT + ncol;
                const float m  = am[v] + bm_s;
                const float lv = al[v] + bl_s;
                const float e  = eps[idx];
                const float zz = m + __expf(0.5f * lv) * e;
                z_out[idx]      = zz;
                mean_out[idx]   = m;
                logvar_out[idx] = lv;
            }
        }
    }
}

extern "C" void kernel_launch(void* const* d_in, const int* in_sizes, int n_in,
                              void* d_out, int out_size, void* d_ws, size_t ws_size,
                              hipStream_t stream) {
    const float* x        = (const float*)d_in[0];
    const float* eps      = (const float*)d_in[1];
    const float* W_emb    = (const float*)d_in[2];
    const float* b_emb    = (const float*)d_in[3];
    const float* W_mean   = (const float*)d_in[4];
    const float* b_mean   = (const float*)d_in[5];
    const float* W_logvar = (const float*)d_in[6];
    const float* b_logvar = (const float*)d_in[7];

    const int batch = in_sizes[0] / D_IN;        // 524288
    float* out = (float*)d_out;
    const long BN = (long)batch * D_OUT;
    float* z_out      = out;
    float* mean_out   = out + BN;
    float* logvar_out = out + 2 * BN;

    dim3 grid(batch / ROWS_PER_BLOCK);           // 2048 blocks
    dim3 block(256);                             // 8 wave32s
    GaussianMLP_vae_enc_kernel<<<grid, block, 0, stream>>>(
        x, eps, W_emb, b_emb, W_mean, b_mean, W_logvar, b_logvar,
        z_out, mean_out, logvar_out, batch);
}
